// PBSAGE_50843822850084
// MI455X (gfx1250) — matrix operand order repaired and checked by reading, not compiled, not run on hardware
//
#include <hip/hip_runtime.h>
#include <hip/hip_bf16.h>

#define N_NODES 50000
#define N_EDGES 400000

typedef __attribute__((ext_vector_type(16))) __bf16 v16bf;
typedef __attribute__((ext_vector_type(8)))  float  v8f;

union FragBF {
  uint4 u[2];
  v16bf v;
};

// ---------------- weight fp32 -> bf16 transpose ([K,N] -> [N][K]) ----------------
__global__ void transpose_bf16_kernel(const float* __restrict__ src,
                                      __hip_bfloat16* __restrict__ dst,
                                      int K, int N) {
  int t = blockIdx.x * 256 + threadIdx.x;
  if (t >= K * N) return;
  int k = t / N, n = t - k * N;
  dst[(size_t)n * K + k] = __float2bfloat16(src[t]);
}

// ---------------- encoder layer 1: [N,24] @ [24,128] + b, ReLU -> bf16 ----------------
__global__ __launch_bounds__(128)
void enc1_kernel(const float* __restrict__ x, const float* __restrict__ w1,
                 const float* __restrict__ b1, __hip_bfloat16* __restrict__ h1) {
  __shared__ float xs[24];
  int node = blockIdx.x;
  int n = threadIdx.x;
  if (n < 24) xs[n] = x[(size_t)node * 24 + n];
  __syncthreads();
  float s = b1[n];
#pragma unroll
  for (int k = 0; k < 24; ++k) s += xs[k] * w1[k * 128 + n];
  h1[(size_t)node * 128 + n] = __float2bfloat16(fmaxf(s, 0.f));
}

// ---------------- zero fill (float4) ----------------
__global__ void zero_kernel(float4* __restrict__ p, int n4) {
  int t = blockIdx.x * 256 + threadIdx.x;
  if (t < n4) p[t] = make_float4(0.f, 0.f, 0.f, 0.f);
}

// ---------------- edge scatter-add: agg[dst] += h[src] (fp32 atomics, bf16 source) ----
__global__ __launch_bounds__(256)
void scatter_kernel(const long long* __restrict__ srcI, const long long* __restrict__ dstI,
                    const __hip_bfloat16* __restrict__ h, float* __restrict__ agg,
                    float* __restrict__ cnt) {
  int t = blockIdx.x * 256 + threadIdx.x;
  int e = t >> 5, g = t & 31;   // 32 threads per edge, 8 features each
  if (e >= N_EDGES) return;
  int s = (int)srcI[e], d = (int)dstI[e];
  uint4 v = *(const uint4*)(h + (size_t)s * 256 + g * 8);
  float* out = agg + (size_t)d * 256 + g * 8;
  const unsigned* w = (const unsigned*)&v;
#pragma unroll
  for (int i = 0; i < 4; ++i) {
    atomicAdd(out + 2 * i,     __builtin_bit_cast(float, w[i] << 16));
    atomicAdd(out + 2 * i + 1, __builtin_bit_cast(float, w[i] & 0xffff0000u));
  }
  if (g == 0) atomicAdd(cnt + d, 1.0f);
}

// ---------------- mean = agg / max(cnt,1), cvt to bf16 ----------------
__global__ void mean_kernel(const float* __restrict__ agg, const float* __restrict__ cnt,
                            __hip_bfloat16* __restrict__ mean) {
  int t = blockIdx.x * 256 + threadIdx.x;  // one thread = 4 features
  if (t >= N_NODES * 64) return;
  int row = t >> 6;
  float inv = 1.0f / fmaxf(cnt[row], 1.0f);
  float4 v = *(const float4*)(agg + (size_t)t * 4);
  __hip_bfloat16* o = mean + (size_t)t * 4;
  o[0] = __float2bfloat16(v.x * inv);
  o[1] = __float2bfloat16(v.y * inv);
  o[2] = __float2bfloat16(v.z * inv);
  o[3] = __float2bfloat16(v.w * inv);
}

// cooperative copy of a 64-column x K panel (col-major [col][K]) into LDS, stride K+8
template <int K>
__device__ inline void copy_panel(const __hip_bfloat16* __restrict__ Wt, int col0,
                                  __hip_bfloat16* dst, int tid) {
  constexpr int CH = K / 8;  // uint4 chunks per column
  for (int idx = tid; idx < 64 * CH; idx += 256) {
    int c = idx / CH, kk = (idx - c * CH) * 8;
    *(uint4*)(dst + c * (K + 8) + kk) =
        *(const uint4*)(Wt + (size_t)(col0 + c) * K + kk);
  }
}

// ---------------- WMMA GEMM: D = relu(A1@W1 [+ A2@W2] + bias), bf16 in, f32 acc ------
// Block = 8 waves; wave w covers rows [row0, row0+16), 64 columns via 4 16x16 tiles.
// B panels staged in LDS (padded stride K+8 to spread banks); A fragment register
// double-buffered so next k-step global loads overlap the 4-WMMA batch.
// Fragment layouts per CDNA5 ISA 7.12.2 (wave32):
//   A (16x32 bf16): lane<16 -> M=lane, K in {kb..kb+7, kb+16..kb+23};
//                   lane>=16 -> M=lane-16, K in {kb+8..kb+15, kb+24..kb+31}.
//   B (32x16 bf16): lane&15 -> column, K split as for A (same K order).
//   C (16x16 f32):  VGPR j -> M = j + 8*(lane>=16), N = lane&15.
template <int K1, int K2, int NOUT>
__global__ __launch_bounds__(256)
void wmma_gemm_kernel(const __hip_bfloat16* __restrict__ A1,
                      const __hip_bfloat16* __restrict__ W1t,   // [NOUT][K1]
                      const __hip_bfloat16* __restrict__ A2,
                      const __hip_bfloat16* __restrict__ W2t,   // [NOUT][K2]
                      const float* __restrict__ bias,
                      __hip_bfloat16* __restrict__ Out, int nrows) {
  extern __shared__ __hip_bfloat16 smem[];
  __hip_bfloat16* p1 = smem;                       // (K1+8)*64
  __hip_bfloat16* p2 = smem + (K1 + 8) * 64;       // (K2+8)*64 (if K2>0)

  const int tid  = threadIdx.x;
  const int lane = tid & 31;
  const int wave = tid >> 5;
  const int row0 = blockIdx.x * 128 + wave * 16;
  const int col0 = blockIdx.y * 64;
  const int half = lane >> 4;
  const int l16  = lane & 15;
  // clamp so out-of-range waves still execute uniformly (stores are guarded)
  const int arowi = min(row0 + l16, nrows - 1);

  copy_panel<K1>(W1t, col0, p1, tid);
  if constexpr (K2 > 0) copy_panel<K2>(W2t, col0, p2, tid);

  v8f acc[4];
#pragma unroll
  for (int t = 0; t < 4; ++t)
#pragma unroll
    for (int j = 0; j < 8; ++j) acc[t][j] = 0.f;

  // issue first A fragment load before the barrier to overlap
  const __hip_bfloat16* arow1 = A1 + (size_t)arowi * K1;
  FragBF a, an;
  a.u[0] = *(const uint4*)(arow1 + half * 8);
  a.u[1] = *(const uint4*)(arow1 + 16 + half * 8);

  __syncthreads();

  for (int kb = 0; kb < K1; kb += 32) {
    if (kb + 32 < K1) {  // prefetch next A fragment
      an.u[0] = *(const uint4*)(arow1 + kb + 32 + half * 8);
      an.u[1] = *(const uint4*)(arow1 + kb + 48 + half * 8);
    }
#pragma unroll
    for (int t = 0; t < 4; ++t) {
      const __hip_bfloat16* bp = p1 + (t * 16 + l16) * (K1 + 8) + kb + half * 8;
      FragBF b;
      b.u[0] = *(const uint4*)bp;
      b.u[1] = *(const uint4*)(bp + 16);
      acc[t] = __builtin_amdgcn_wmma_f32_16x16x32_bf16(
          false, a.v, false, b.v, (short)0, acc[t], false, false);
    }
    a = an;
  }

  if constexpr (K2 > 0) {
    const __hip_bfloat16* arow2 = A2 + (size_t)arowi * K2;
    a.u[0] = *(const uint4*)(arow2 + half * 8);
    a.u[1] = *(const uint4*)(arow2 + 16 + half * 8);
    for (int kb = 0; kb < K2; kb += 32) {
      if (kb + 32 < K2) {
        an.u[0] = *(const uint4*)(arow2 + kb + 32 + half * 8);
        an.u[1] = *(const uint4*)(arow2 + kb + 48 + half * 8);
      }
#pragma unroll
      for (int t = 0; t < 4; ++t) {
        const __hip_bfloat16* bp = p2 + (t * 16 + l16) * (K2 + 8) + kb + half * 8;
        FragBF b;
        b.u[0] = *(const uint4*)bp;
        b.u[1] = *(const uint4*)(bp + 16);
        acc[t] = __builtin_amdgcn_wmma_f32_16x16x32_bf16(
            false, a.v, false, b.v, (short)0, acc[t], false, false);
      }
      a = an;
    }
  }

  if (row0 < nrows) {
#pragma unroll
    for (int t = 0; t < 4; ++t) {
      const int colg = col0 + t * 16 + l16;
      const float bv = bias[colg];
#pragma unroll
      for (int j = 0; j < 8; ++j) {
        const int rowg = row0 + half * 8 + j;
        float vv = fmaxf(acc[t][j] + bv, 0.f);
        Out[(size_t)rowg * NOUT + colg] = __float2bfloat16(vv);
      }
    }
  }
}

// ---------------- readout layer 2: [N,128] @ [128,12] + b -> f32 out ----------------
__global__ __launch_bounds__(192)
void ro2_kernel(const __hip_bfloat16* __restrict__ r1, const float* __restrict__ w2,
                const float* __restrict__ b2, float* __restrict__ out) {
  __shared__ float wsm[128 * 12];
  for (int i = threadIdx.x; i < 128 * 12; i += 192) wsm[i] = w2[i];
  __syncthreads();
  int node = blockIdx.x * 16 + threadIdx.x / 12;   // 192 = 16 nodes x 12 outputs
  int o = threadIdx.x % 12;
  if (node >= N_NODES) return;
  const __hip_bfloat16* row = r1 + (size_t)node * 128;
  float s = b2[o];
#pragma unroll 4
  for (int k = 0; k < 128; ++k) s += __bfloat162float(row[k]) * wsm[k * 12 + o];
  out[(size_t)node * 12 + o] = s;
}

extern "C" void kernel_launch(void* const* d_in, const int* in_sizes, int n_in,
                              void* d_out, int out_size, void* d_ws, size_t ws_size,
                              hipStream_t stream) {
  const float*     x      = (const float*)d_in[0];
  const long long* ei     = (const long long*)d_in[1];   // int64 edge_index [2,E]
  const float*     enc_w1 = (const float*)d_in[2];
  const float*     enc_b1 = (const float*)d_in[3];
  const float*     enc_w2 = (const float*)d_in[4];
  const float*     enc_b2 = (const float*)d_in[5];
  const float*     s1_wl  = (const float*)d_in[6];
  const float*     s1_bl  = (const float*)d_in[7];
  const float*     s1_wr  = (const float*)d_in[8];
  const float*     s2_wl  = (const float*)d_in[9];
  const float*     s2_bl  = (const float*)d_in[10];
  const float*     s2_wr  = (const float*)d_in[11];
  const float*     ro_w1  = (const float*)d_in[12];
  const float*     ro_b1  = (const float*)d_in[13];
  const float*     ro_w2  = (const float*)d_in[14];
  const float*     ro_b2  = (const float*)d_in[15];
  const long long* e_src  = ei;
  const long long* e_dst  = ei + N_EDGES;
  float* outp = (float*)d_out;
  (void)in_sizes; (void)n_in; (void)out_size; (void)ws_size;

  // ---- workspace bump allocator (256B aligned slots) ----
  char* wsp = (char*)d_ws;
  auto alloc = [&](size_t bytes) {
    void* p = (void*)wsp;
    wsp += (bytes + 255) & ~(size_t)255;
    return p;
  };
  float* agg = (float*)alloc((size_t)N_NODES * 256 * sizeof(float)); // 51.2MB (256-mult)
  float* cnt = (float*)alloc((size_t)N_NODES * sizeof(float));       // directly follows agg
  __hip_bfloat16* hA   = (__hip_bfloat16*)alloc((size_t)N_NODES * 256 * 2);
  __hip_bfloat16* hB   = (__hip_bfloat16*)alloc((size_t)N_NODES * 256 * 2);
  __hip_bfloat16* sbuf = (__hip_bfloat16*)alloc((size_t)N_NODES * 256 * 2); // h1/mean/r1
  __hip_bfloat16* enc_w2t = (__hip_bfloat16*)alloc(128 * 256 * 2);
  __hip_bfloat16* s1_wlt  = (__hip_bfloat16*)alloc(256 * 256 * 2);
  __hip_bfloat16* s1_wrt  = (__hip_bfloat16*)alloc(256 * 256 * 2);
  __hip_bfloat16* s2_wlt  = (__hip_bfloat16*)alloc(256 * 256 * 2);
  __hip_bfloat16* s2_wrt  = (__hip_bfloat16*)alloc(256 * 256 * 2);
  __hip_bfloat16* ro_w1t  = (__hip_bfloat16*)alloc(256 * 128 * 2);

  // ---- weight conversion (every call; deterministic) ----
  transpose_bf16_kernel<<<(128 * 256 + 255) / 256, 256, 0, stream>>>(enc_w2, enc_w2t, 128, 256);
  transpose_bf16_kernel<<<(256 * 256 + 255) / 256, 256, 0, stream>>>(s1_wl, s1_wlt, 256, 256);
  transpose_bf16_kernel<<<(256 * 256 + 255) / 256, 256, 0, stream>>>(s1_wr, s1_wrt, 256, 256);
  transpose_bf16_kernel<<<(256 * 256 + 255) / 256, 256, 0, stream>>>(s2_wl, s2_wlt, 256, 256);
  transpose_bf16_kernel<<<(256 * 256 + 255) / 256, 256, 0, stream>>>(s2_wr, s2_wrt, 256, 256);
  transpose_bf16_kernel<<<(256 * 128 + 255) / 256, 256, 0, stream>>>(ro_w1, ro_w1t, 256, 128);

  const dim3 g256((N_NODES + 127) / 128, 4);  // NOUT=256 -> 4 column blocks of 64
  const dim3 g128((N_NODES + 127) / 128, 2);  // NOUT=128 -> 2 column blocks of 64
  const int zero_n4 = (N_NODES * 256 + N_NODES) / 4;  // agg + cnt contiguous
  // dynamic LDS: (K+8)-stride bf16 panels of 64 columns
  const size_t sh_enc2 = (size_t)(128 + 8) * 64 * 2;                      // 17.4 KB
  const size_t sh_sage = ((size_t)(256 + 8) + (256 + 8)) * 64 * 2;        // 67.6 KB
  const size_t sh_ro1  = (size_t)(256 + 8) * 64 * 2;                      // 33.8 KB

  // ---- encoder ----
  enc1_kernel<<<N_NODES, 128, 0, stream>>>(x, enc_w1, enc_b1, sbuf);
  wmma_gemm_kernel<128, 0, 256><<<g256, 256, sh_enc2, stream>>>(
      sbuf, enc_w2t, nullptr, nullptr, enc_b2, hA, N_NODES);

  // ---- SAGE layer 1 ----
  zero_kernel<<<(zero_n4 + 255) / 256, 256, 0, stream>>>((float4*)agg, zero_n4);
  scatter_kernel<<<(N_EDGES * 32) / 256, 256, 0, stream>>>(e_src, e_dst, hA, agg, cnt);
  mean_kernel<<<(N_NODES * 64 + 255) / 256, 256, 0, stream>>>(agg, cnt, sbuf);
  wmma_gemm_kernel<256, 256, 256><<<g256, 256, sh_sage, stream>>>(
      sbuf, s1_wlt, hA, s1_wrt, s1_bl, hB, N_NODES);

  // ---- SAGE layer 2 ----
  zero_kernel<<<(zero_n4 + 255) / 256, 256, 0, stream>>>((float4*)agg, zero_n4);
  scatter_kernel<<<(N_EDGES * 32) / 256, 256, 0, stream>>>(e_src, e_dst, hB, agg, cnt);
  mean_kernel<<<(N_NODES * 64 + 255) / 256, 256, 0, stream>>>(agg, cnt, sbuf);
  wmma_gemm_kernel<256, 256, 256><<<g256, 256, sh_sage, stream>>>(
      sbuf, s2_wlt, hB, s2_wrt, s2_bl, hA, N_NODES);

  // ---- readout ----
  wmma_gemm_kernel<256, 0, 128><<<g128, 256, sh_ro1, stream>>>(
      hA, ro_w1t, nullptr, nullptr, ro_b1, sbuf, N_NODES);
  ro2_kernel<<<(N_NODES + 15) / 16, 192, 0, stream>>>(sbuf, ro_w2, ro_b2, outp);
}
